// SelfAttention_7825430413323
// MI455X (gfx1250) — matrix operand order repaired and checked by reading, not compiled, not run on hardware
//
#include <hip/hip_runtime.h>

// ---------------- problem constants ----------------
#define BB   8
#define CC   512
#define NN   2048
#define CQK  64      // C/8

typedef __attribute__((ext_vector_type(16))) __bf16 v16bf;
typedef __attribute__((ext_vector_type(8)))  __bf16 v8bf;
typedef __attribute__((ext_vector_type(8)))  float  v8f;
typedef __attribute__((ext_vector_type(4)))  int    i4v;

// ---- optional CDNA5 async global->LDS path (ASYNCcnt), with sync fallback ----
#if defined(__has_builtin)
#if __has_builtin(__builtin_amdgcn_global_load_async_to_lds_b128) && \
    __has_builtin(__builtin_amdgcn_s_wait_asynccnt)
#define HAVE_ASYNC_LDS 1
#endif
#endif

#if defined(HAVE_ASYNC_LDS)
typedef __attribute__((address_space(1))) i4v* g_i4p;   // global int4*
typedef __attribute__((address_space(3))) i4v* l_i4p;   // LDS int4*
#define ASYNC_COPY16(srcp, dstp)                                               \
    __builtin_amdgcn_global_load_async_to_lds_b128(                            \
        (g_i4p)(srcp), (l_i4p)(dstp), 0, 0)
#define ASYNC_WAIT() __builtin_amdgcn_s_wait_asynccnt(0)
#else
#define ASYNC_WAIT()
#endif

// f32 -> bf16 round-to-nearest-even (bit twiddling; avoids relying on __bf16 arith)
__device__ inline __bf16 f2bf(float f) {
    union { float f; unsigned u; } c; c.f = f;
    unsigned r = (c.u + 0x7FFFu + ((c.u >> 16) & 1u)) >> 16;
    unsigned short h = (unsigned short)r;
    return __builtin_bit_cast(__bf16, h);
}

__device__ inline v8f wmma_bf16(v16bf a, v16bf b, v8f c) {
    // (neg_a, A, neg_b, B, c_mod, C, reuse_a, reuse_b)
    return __builtin_amdgcn_wmma_f32_16x16x32_bf16(false, a, false, b, (short)0, c, false, false);
}

// A fragment: 16x32 bf16, source row-major [M x K], K contiguous.
// lanes 0-15 : M=lane,    K = k0+0..15
// lanes 16-31: M=lane-16, K = k0+16..31
__device__ inline v16bf load_frag_A(const __bf16* base, int lda, int m0, int k0, int lane) {
    int m  = lane & 15;
    int ko = (lane >> 4) << 4;                 // 0 or 16
    const __bf16* p = base + (size_t)(m0 + m) * lda + k0 + ko;
    v8bf lo = *(const v8bf*)(p);
    v8bf hi = *(const v8bf*)(p + 8);
    v16bf a;
#pragma unroll
    for (int t = 0; t < 8; ++t) { a[t] = lo[t]; a[t + 8] = hi[t]; }
    return a;
}

// B fragment: 32x16 bf16, source stored as B^T row-major [N x K] (column contiguous in K).
// lanes 0-15 : N=lane, K = k0+0..15 ; lanes 16-31: N=lane-16, K = k0+16..31
__device__ inline v16bf load_frag_B(const __bf16* baseT, int ldb, int n0, int k0, int lane) {
    int n  = lane & 15;
    int ko = (lane >> 4) << 4;                 // 0 or 16
    return *(const v16bf*)(baseT + (size_t)(n0 + n) * ldb + k0 + ko);
}

// ---------------- kernel 1a: x [B,C,N] f32 -> xT [B,N,C] bf16 ----------------
__global__ __launch_bounds__(256) void k_transpose_cvt_x(const float* __restrict__ x,
                                                         __bf16* __restrict__ xT) {
    size_t id = (size_t)blockIdx.x * blockDim.x + threadIdx.x;
    if (id >= (size_t)BB * CC * NN) return;
    int n = (int)(id % NN);
    int c = (int)((id / NN) % CC);
    int b = (int)(id / ((size_t)CC * NN));
    xT[((size_t)b * NN + n) * CC + c] = f2bf(x[id]);
}

// ---------------- kernel 1b: generic f32 -> bf16 ----------------
__global__ __launch_bounds__(256) void k_cvt_bf16(const float* __restrict__ s,
                                                  __bf16* __restrict__ d, int n) {
    int id = blockIdx.x * blockDim.x + threadIdx.x;
    if (id < n) d[id] = f2bf(s[id]);
}

// ---------------- kernel 2: qT,kT = xT * W^T  (M=i, K=c, N=d) ----------------
__global__ __launch_bounds__(256) void k_qk_proj(const __bf16* __restrict__ xT,
                                                 const __bf16* __restrict__ wq,
                                                 const __bf16* __restrict__ wk,
                                                 const float* __restrict__ bq,
                                                 const float* __restrict__ bk,
                                                 __bf16* __restrict__ qT,
                                                 __bf16* __restrict__ kT) {
    int lane = threadIdx.x & 31;
    int wave = threadIdx.x >> 5;
    int tile = blockIdx.x * (blockDim.x >> 5) + wave;   // B * (N/16) * (CQK/16) * 2
    int which = tile & 1;  tile >>= 1;
    int b  = tile / ((NN / 16) * (CQK / 16));
    int r  = tile % ((NN / 16) * (CQK / 16));
    int i0 = (r / (CQK / 16)) * 16;
    int d0 = (r % (CQK / 16)) * 16;

    const __bf16* w    = which ? wk : wq;
    const float*  bias = which ? bk : bq;
    __bf16*       dst  = (which ? kT : qT) + (size_t)b * NN * CQK;
    const __bf16* xb   = xT + (size_t)b * NN * CC;

    v8f acc = {};
#pragma unroll 4
    for (int k0 = 0; k0 < CC; k0 += 32) {
        v16bf a  = load_frag_A(xb, CC, i0, k0, lane);   // rows of xT (i x c)
        v16bf bf = load_frag_B(w,  CC, d0, k0, lane);   // col d of W^T == row d of W
        acc = wmma_bf16(a, bf, acc);
    }
    int n  = lane & 15;
    int mh = (lane >> 4) * 8;
    float bv = bias[d0 + n];
#pragma unroll
    for (int rr = 0; rr < 8; ++rr)
        dst[(size_t)(i0 + mh + rr) * CQK + d0 + n] = f2bf(acc[rr] + bv);
}

// ---------------- kernel 3: v = Wv * x  (M=c, K=c', N=i) -> v_bf [B,C,N] ----------------
__global__ __launch_bounds__(256) void k_v_proj(const __bf16* __restrict__ xT,
                                                const __bf16* __restrict__ wv,
                                                const float* __restrict__ bv,
                                                __bf16* __restrict__ v_bf) {
    int lane = threadIdx.x & 31;
    int wave = threadIdx.x >> 5;
    int tile = blockIdx.x * (blockDim.x >> 5) + wave;   // B * (C/16) * (N/16)
    int b  = tile / ((CC / 16) * (NN / 16));
    int r  = tile % ((CC / 16) * (NN / 16));
    int c0 = (r / (NN / 16)) * 16;
    int n0 = (r % (NN / 16)) * 16;

    const __bf16* xb = xT + (size_t)b * NN * CC;
    __bf16* dst = v_bf + (size_t)b * CC * NN;

    v8f acc = {};
#pragma unroll 4
    for (int k0 = 0; k0 < CC; k0 += 32) {
        v16bf a  = load_frag_A(wv, CC, c0, k0, lane);   // rows of Wv (c x c')
        v16bf bf = load_frag_B(xb, CC, n0, k0, lane);   // col i of x == row i of xT
        acc = wmma_bf16(a, bf, acc);
    }
    int n  = lane & 15;
    int mh = (lane >> 4) * 8;
#pragma unroll
    for (int rr = 0; rr < 8; ++rr)
        dst[(size_t)(c0 + mh + rr) * NN + n0 + n] = f2bf(acc[rr] + bv[c0 + mh + rr]);
}

// ---------------- kernel 4: scores + online softmax stats ----------------
__global__ __launch_bounds__(128) void k_scores(const __bf16* __restrict__ qT,
                                                const __bf16* __restrict__ kT,
                                                float* __restrict__ attn,
                                                float* __restrict__ rowmax,
                                                float* __restrict__ rowsum) {
    int lane  = threadIdx.x & 31;
    int wave  = threadIdx.x >> 5;
    int strip = blockIdx.x * (blockDim.x >> 5) + wave;  // B * N/16
    int b  = strip / (NN / 16);
    int i0 = (strip % (NN / 16)) * 16;

    const __bf16* q = qT + (size_t)b * NN * CQK;
    const __bf16* k = kT + (size_t)b * NN * CQK;
    float* att = attn + (size_t)b * NN * NN;

    v16bf a0 = load_frag_A(q, CQK, i0, 0,  lane);
    v16bf a1 = load_frag_A(q, CQK, i0, 32, lane);

    float m_run[8], s_run[8];
#pragma unroll
    for (int rr = 0; rr < 8; ++rr) { m_run[rr] = -3.4e38f; s_run[rr] = 0.f; }

    int n  = lane & 15;
    int mh = (lane >> 4) * 8;

    for (int j0 = 0; j0 < NN; j0 += 16) {
        v16bf b0 = load_frag_B(k, CQK, j0, 0,  lane);
        v16bf b1 = load_frag_B(k, CQK, j0, 32, lane);
        v8f d = {};
        d = wmma_bf16(a0, b0, d);
        d = wmma_bf16(a1, b1, d);

#pragma unroll
        for (int rr = 0; rr < 8; ++rr)
            att[(size_t)(i0 + mh + rr) * NN + j0 + n] = d[rr];

#pragma unroll
        for (int rr = 0; rr < 8; ++rr) {
            float tmax = d[rr];
#pragma unroll
            for (int off = 1; off < 16; off <<= 1)
                tmax = fmaxf(tmax, __shfl_xor(tmax, off, 32));
            float mn = fmaxf(m_run[rr], tmax);
            float e  = __expf(d[rr] - mn);
#pragma unroll
            for (int off = 1; off < 16; off <<= 1)
                e += __shfl_xor(e, off, 32);
            s_run[rr] = s_run[rr] * __expf(m_run[rr] - mn) + e;
            m_run[rr] = mn;
        }
    }
    if ((lane & 15) == 0) {
#pragma unroll
        for (int rr = 0; rr < 8; ++rr) {
            rowmax[(size_t)b * NN + i0 + mh + rr] = m_run[rr];
            rowsum[(size_t)b * NN + i0 + mh + rr] = s_run[rr];
        }
    }
}

// ---------------- kernel 5: normalize attn in place (float4, L2-resident) ----------------
__global__ __launch_bounds__(256) void k_softmax_norm(float* __restrict__ attn,
                                                      const float* __restrict__ rowmax,
                                                      const float* __restrict__ rowsum) {
    size_t id = (size_t)blockIdx.x * blockDim.x + threadIdx.x;
    size_t base = id * 4;
    if (base >= (size_t)BB * NN * NN) return;
    int b = (int)(base / ((size_t)NN * NN));
    size_t rem = base % ((size_t)NN * NN);
    int i = (int)(rem / NN);
    float m   = rowmax[(size_t)b * NN + i];
    float inv = 1.0f / rowsum[(size_t)b * NN + i];
    float4 v = *(const float4*)(attn + base);
    v.x = __expf(v.x - m) * inv;
    v.y = __expf(v.y - m) * inv;
    v.z = __expf(v.z - m) * inv;
    v.w = __expf(v.w - m) * inv;
    *(float4*)(attn + base) = v;
}

// ---------------- kernel 6: out = gamma * (v @ attn) + x ----------------
// LDS-blocked GEMM: block tile 128(c) x 64(j), K-step 32, double-buffered LDS.
// A (v, bf16) staged via async global->LDS when available; B (attn f32) staged
// with coalesced f32 loads + in-register bf16 convert + transposed ds_store.
#define BM     128
#define BN     64
#define KSTEP  32
#define KPAD   40     // 80B row pitch -> conflict-free 16-lane fragment reads
#define KSTEPS (NN / KSTEP)

__global__ __launch_bounds__(256) void k_attn_v_out(const __bf16* __restrict__ v_bf,
                                                    const float* __restrict__ attn,
                                                    const float* __restrict__ x,
                                                    const float* __restrict__ gamma,
                                                    float* __restrict__ out) {
    __shared__ __align__(16) __bf16 As[2][BM][KPAD];   // v tile     [c][k]
    __shared__ __align__(16) __bf16 Bs[2][BN][KPAD];   // attn^T tile [j][k]

    const int t    = threadIdx.x;
    const int lane = t & 31;
    const int wave = t >> 5;

    int blk = blockIdx.x;                              // B * (C/BM) * (N/BN)
    int b    = blk / ((CC / BM) * (NN / BN));
    int r    = blk % ((CC / BM) * (NN / BN));
    int cblk = (r / (NN / BN)) * BM;
    int jblk = (r % (NN / BN)) * BN;

    const __bf16* vb  = v_bf + (size_t)b * CC * NN;
    const float*  att = attn + (size_t)b * NN * NN;

    // wave grid inside block tile: 4 waves along M, 2 along N; 2x2 frags each
    const int wm = (wave & 3) * 32;
    const int wn = (wave >> 2) * 32;

    v8f acc[2][2] = {};

    // staging maps
    const int arow = t >> 1,  akh = (t & 1) * 16;      // A: 16 bf16 (32B) per thread
    const int bj   = t & 63,  bkg = (t >> 6) * 8;      // B: 8 f32 column loads per thread

    auto stageA = [&](int buf, int i0) {
        const __bf16* g = vb + (size_t)(cblk + arow) * NN + i0 + akh;
#if defined(HAVE_ASYNC_LDS)
        ASYNC_COPY16(g,     &As[buf][arow][akh]);
        ASYNC_COPY16(g + 8, &As[buf][arow][akh + 8]);
#else
        v8bf u0 = *(const v8bf*)g;
        v8bf u1 = *(const v8bf*)(g + 8);
        *(v8bf*)&As[buf][arow][akh]     = u0;
        *(v8bf*)&As[buf][arow][akh + 8] = u1;
#endif
    };
    auto stageB = [&](int buf, int i0) {
        const float* g = att + (size_t)(i0 + bkg) * NN + jblk + bj;
        __builtin_prefetch(g + (size_t)2 * KSTEP * NN, 0, 1);   // global_prefetch ahead
        v8bf pk;
#pragma unroll
        for (int q = 0; q < 8; ++q) pk[q] = f2bf(g[(size_t)q * NN]);
        *(v8bf*)&Bs[buf][bj][bkg] = pk;
    };
    auto compute = [&](int buf) {
        const int m   = lane & 15;
        const int ako = (lane >> 4) * 16;
        const int bko = (lane >> 4) * 16;
        const int n   = lane & 15;
#pragma unroll
        for (int mi = 0; mi < 2; ++mi) {
            const __bf16* ap = &As[buf][wm + mi * 16 + m][ako];
            v8bf alo = *(const v8bf*)ap;
            v8bf ahi = *(const v8bf*)(ap + 8);
            v16bf a;
#pragma unroll
            for (int q = 0; q < 8; ++q) { a[q] = alo[q]; a[q + 8] = ahi[q]; }
#pragma unroll
            for (int ni = 0; ni < 2; ++ni) {
                const __bf16* bp = &Bs[buf][wn + ni * 16 + n][bko];
                v8bf b0 = *(const v8bf*)bp;
                v8bf b1 = *(const v8bf*)(bp + 8);
                v16bf bb;
#pragma unroll
                for (int q = 0; q < 8; ++q) { bb[q] = b0[q]; bb[q + 8] = b1[q]; }
                acc[mi][ni] = wmma_bf16(a, bb, acc[mi][ni]);
            }
        }
    };

    // prologue
    stageA(0, 0);
    stageB(0, 0);
    ASYNC_WAIT();
    __syncthreads();

#pragma unroll 1
    for (int s = 0; s < KSTEPS; ++s) {
        int cur = s & 1;
        if (s + 1 < KSTEPS) {               // issue next-stage loads first (overlap)
            stageA(cur ^ 1, (s + 1) * KSTEP);
            stageB(cur ^ 1, (s + 1) * KSTEP);
        }
        compute(cur);                       // WMMA on current buffer
        ASYNC_WAIT();                       // next buffer's async loads done
        __syncthreads();                    // publish next buffer / retire current
    }

    // epilogue: out = gamma * acc + x
    float g  = *gamma;
    int   n  = lane & 15;
    int   mh = (lane >> 4) * 8;
#pragma unroll
    for (int mi = 0; mi < 2; ++mi)
#pragma unroll
        for (int ni = 0; ni < 2; ++ni)
#pragma unroll
            for (int rr = 0; rr < 8; ++rr) {
                int cc = cblk + wm + mi * 16 + mh + rr;
                int jj = jblk + wn + ni * 16 + n;
                size_t idx = (size_t)b * CC * NN + (size_t)cc * NN + jj;
                out[idx] = g * acc[mi][ni][rr] + x[idx];
            }
}

// ---------------- host launcher ----------------
extern "C" void kernel_launch(void* const* d_in, const int* in_sizes, int n_in,
                              void* d_out, int out_size, void* d_ws, size_t ws_size,
                              hipStream_t stream) {
    const float* x     = (const float*)d_in[0];
    const float* Wq    = (const float*)d_in[1];
    const float* bq    = (const float*)d_in[2];
    const float* Wk    = (const float*)d_in[3];
    const float* bk    = (const float*)d_in[4];
    const float* Wv    = (const float*)d_in[5];
    const float* bv    = (const float*)d_in[6];
    const float* gamma = (const float*)d_in[7];

    float* out  = (float*)d_out;                                   // [B,C,N]
    float* attn = (float*)d_out + (size_t)BB * CC * NN;            // [B,N,N]

    char* ws = (char*)d_ws;
    size_t off = 0;
    __bf16* xT   = (__bf16*)(ws + off); off += (size_t)BB * NN * CC * 2;   // 16 MB
    __bf16* wqb  = (__bf16*)(ws + off); off += (size_t)CQK * CC * 2;
    __bf16* wkb  = (__bf16*)(ws + off); off += (size_t)CQK * CC * 2;
    __bf16* wvb  = (__bf16*)(ws + off); off += (size_t)CC * CC * 2;
    __bf16* qT   = (__bf16*)(ws + off); off += (size_t)BB * NN * CQK * 2;
    __bf16* kT   = (__bf16*)(ws + off); off += (size_t)BB * NN * CQK * 2;
    __bf16* vbf  = (__bf16*)(ws + off); off += (size_t)BB * CC * NN * 2;   // 16 MB
    float*  rmax = (float*)(ws + off);  off += (size_t)BB * NN * 4;
    float*  rsum = (float*)(ws + off);  off += (size_t)BB * NN * 4;

    {
        size_t tot = (size_t)BB * CC * NN;
        k_transpose_cvt_x<<<(unsigned)((tot + 255) / 256), 256, 0, stream>>>(x, xT);
        k_cvt_bf16<<<(CQK * CC + 255) / 256, 256, 0, stream>>>(Wq, wqb, CQK * CC);
        k_cvt_bf16<<<(CQK * CC + 255) / 256, 256, 0, stream>>>(Wk, wkb, CQK * CC);
        k_cvt_bf16<<<(CC * CC + 255) / 256, 256, 0, stream>>>(Wv, wvb, CC * CC);
    }
    k_qk_proj<<<1024, 256, 0, stream>>>(xT, wqb, wkb, bq, bk, qT, kT);
    k_v_proj<<<4096, 256, 0, stream>>>(xT, wvb, bv, vbf);
    k_scores<<<256, 128, 0, stream>>>(qT, kT, attn, rmax, rsum);
    k_softmax_norm<<<32768, 256, 0, stream>>>(attn, rmax, rsum);
    // 8*(512/128)*(2048/64) = 1024 blocks, 256 threads (8 waves)
    k_attn_v_out<<<1024, 256, 0, stream>>>(vbf, attn, x, gamma, out);
}
